// GAT_70970039599316
// MI455X (gfx1250) — matrix operand order repaired and checked by reading, not compiled, not run on hardware
//
#include <hip/hip_runtime.h>

#define N_NODES 2048
#define NFEAT   512
#define NHID    64
#define NCLASS  64
#define NHEADS  8
#define ALPHA   0.2f

typedef __attribute__((ext_vector_type(16))) _Float16 v16h;
typedef __attribute__((ext_vector_type(8)))  _Float16 v8h;
typedef __attribute__((ext_vector_type(8)))  float    v8f;

__device__ __forceinline__ float lrelu(float x) { return x > 0.f ? x : ALPHA * x; }

// ---------------------------------------------------------------------------
// Elementwise helpers
// ---------------------------------------------------------------------------
__global__ void cvt_f32_to_f16(const float* __restrict__ src, _Float16* __restrict__ dst, int n) {
  for (int i = blockIdx.x * blockDim.x + threadIdx.x; i < n; i += gridDim.x * blockDim.x)
    dst[i] = (_Float16)src[i];
}

__global__ void elu_f16(const _Float16* __restrict__ src, _Float16* __restrict__ dst, int n) {
  for (int i = blockIdx.x * blockDim.x + threadIdx.x; i < n; i += gridDim.x * blockDim.x) {
    float v = (float)src[i];
    dst[i] = (_Float16)(v > 0.f ? v : (__expf(v) - 1.f));
  }
}

// ---------------------------------------------------------------------------
// WMMA fragment loaders (CDNA5 16x16x32 f16 layouts, wave32)
// A 16x32: lane L -> row M=L%16; v16h elems 0..7 = K 8*(L/16)..+7,
//          elems 8..15 = K 16+8*(L/16)..+7
// B 32x16: lane L -> N=L%16; v16h elems = K same as A
// D 16x16 f32: c[r] -> (M = r + 8*(L>>4), N = L&15)
// ---------------------------------------------------------------------------
__device__ __forceinline__ v16h load_A_frag(const _Float16* rowPtr, int halfl) {
  v8h lo = *(const v8h*)(rowPtr + 8 * halfl);
  v8h hi = *(const v8h*)(rowPtr + 16 + 8 * halfl);
  v16h a;
#pragma unroll
  for (int i = 0; i < 8; ++i) { a[i] = lo[i]; a[8 + i] = hi[i]; }
  return a;
}

__device__ __forceinline__ v16h load_B_frag(const _Float16* colPtr, int halfl) {
  v16h b;
#pragma unroll
  for (int i = 0; i < 8; ++i) {
    b[i] = colPtr[(8 * halfl + i) * 72];
    b[8 + i] = colPtr[(16 + 8 * halfl + i) * 72];
  }
  return b;
}

// ---------------------------------------------------------------------------
// GEMM: C[head] = A(MxK, row-major f16, ldA) @ B[head](Kx64, row-major f16)
// C f16, ld 64. Block = 256 threads (8 waves), tile 128 rows x 64 cols.
// ---------------------------------------------------------------------------
__global__ __launch_bounds__(256)
void gemm_wmma(const _Float16* __restrict__ A, int ldA, int K,
               const _Float16* __restrict__ Bbase, long bHeadStride,
               _Float16* __restrict__ Cbase, long cHeadStride) {
  __shared__ _Float16 As[128][48];  // 32 cols + pad (96B row stride, 16B aligned)
  __shared__ _Float16 Bs[32][72];   // 64 cols + pad (144B row stride)

  const int tid  = threadIdx.x;
  const int wave = tid >> 5, lane = tid & 31;
  const int mrow = lane & 15, halfl = lane >> 4;
  const int m0   = blockIdx.x * 128;
  const int head = blockIdx.y;
  const _Float16* B = Bbase + (size_t)head * bHeadStride;
  _Float16*       C = Cbase + (size_t)head * cHeadStride;

  v8f acc[4] = {};

  for (int k0 = 0; k0 < K; k0 += 32) {
    // Stage A tile 128x32: 512 16B chunks, 2 per thread
#pragma unroll
    for (int it = 0; it < 2; ++it) {
      int c = tid + it * 256;
      int r = c >> 2, col = (c & 3) * 8;
      *(v8h*)&As[r][col] = *(const v8h*)(A + (size_t)(m0 + r) * ldA + k0 + col);
    }
    // Stage B tile 32x64: 256 16B chunks, 1 per thread
    {
      int r = tid >> 3, col = (tid & 7) * 8;
      *(v8h*)&Bs[r][col] = *(const v8h*)(B + (size_t)(k0 + r) * 64 + col);
    }
    __syncthreads();

    v16h a = load_A_frag(&As[wave * 16 + mrow][0], halfl);
#pragma unroll
    for (int nt = 0; nt < 4; ++nt) {
      v16h b = load_B_frag(&Bs[0][nt * 16 + mrow], halfl);
      acc[nt] = __builtin_amdgcn_wmma_f32_16x16x32_f16(
          false, a, false, b, (short)0, acc[nt], false, false);
    }
    __syncthreads();
  }

#pragma unroll
  for (int nt = 0; nt < 4; ++nt)
#pragma unroll
    for (int r = 0; r < 8; ++r)
      C[(size_t)(m0 + wave * 16 + 8 * halfl + r) * 64 + nt * 16 + mrow] =
          (_Float16)acc[nt][r];
}

// ---------------------------------------------------------------------------
// Per-(node, head) attention projections: s1 = h.a1, s2 = h.a2 (d = 64)
// ---------------------------------------------------------------------------
__global__ __launch_bounds__(64)
void scores_kernel(const _Float16* __restrict__ h, const float* __restrict__ a1,
                   const float* __restrict__ a2, float* __restrict__ s1,
                   float* __restrict__ s2) {
  const int n = blockIdx.x, head = blockIdx.y, t = threadIdx.x;
  const _Float16* hp = h + ((size_t)head * N_NODES + n) * NHID;
  float v  = (float)hp[t];
  __shared__ float r1[64], r2[64];
  r1[t] = v * a1[head * NHID + t];
  r2[t] = v * a2[head * NHID + t];
  __syncthreads();
  for (int s = 32; s > 0; s >>= 1) {
    if (t < s) { r1[t] += r1[t + s]; r2[t] += r2[t + s]; }
    __syncthreads();
  }
  if (t == 0) {
    s1[(size_t)head * N_NODES + n] = r1[0];
    s2[(size_t)head * N_NODES + n] = r2[0];
  }
}

// ---------------------------------------------------------------------------
// Row stats: M_i = lrelu(max_masked(s1) + s2_i); dinv_i = 1/sum exp(e - M_i)
// One wave32 per (row, head). lrelu monotone => max commutes with it.
// ---------------------------------------------------------------------------
__global__ __launch_bounds__(32)
void rowstats_kernel(const int* __restrict__ adj, const float* __restrict__ s1,
                     const float* __restrict__ s2, float* __restrict__ Mrow,
                     float* __restrict__ dinv) {
  const int i = blockIdx.x, head = blockIdx.y, lane = threadIdx.x;
  const int*   arow = adj + (size_t)i * N_NODES;
  const float* s1h  = s1 + (size_t)head * N_NODES;

  float m = -1e30f;
  for (int j = lane; j < N_NODES; j += 32)
    if (arow[j] > 0) m = fmaxf(m, s1h[j]);
#pragma unroll
  for (int off = 16; off > 0; off >>= 1) m = fmaxf(m, __shfl_xor(m, off, 32));

  const float s2i = s2[(size_t)head * N_NODES + i];
  const float Mi  = lrelu(m + s2i);

  float d = 0.f;
  for (int j = lane; j < N_NODES; j += 32)
    if (arow[j] > 0) d += __expf(lrelu(s1h[j] + s2i) - Mi);
#pragma unroll
  for (int off = 16; off > 0; off >>= 1) d += __shfl_xor(d, off, 32);

  if (lane == 0) {
    Mrow[(size_t)head * N_NODES + i] = Mi;
    dinv[(size_t)head * N_NODES + i] = 1.f / d;
  }
}

// ---------------------------------------------------------------------------
// Attention aggregation: out = diag(dinv) * P @ V, with P tiles regenerated
// on the fly into LDS (never materialized in HBM). Tile 128 rows x 64 cols.
// ---------------------------------------------------------------------------
template <bool F32OUT>
__global__ __launch_bounds__(256)
void agg_wmma(const int* __restrict__ adj, const float* __restrict__ s1,
              const float* __restrict__ s2, const float* __restrict__ Mrow,
              const float* __restrict__ dinv, const _Float16* __restrict__ Vbase,
              long vHeadStride, void* __restrict__ outv, int ldOut,
              int colHeadStride) {
  __shared__ _Float16 Ps[128][48];
  __shared__ _Float16 Vs[32][72];
  __shared__ float s2s[128], Ms[128], ds[128];

  const int tid  = threadIdx.x;
  const int wave = tid >> 5, lane = tid & 31;
  const int mrow = lane & 15, halfl = lane >> 4;
  const int i0   = blockIdx.x * 128;
  const int head = blockIdx.y;

  const float* s1h = s1 + (size_t)head * N_NODES;
  const float* s2h = s2 + (size_t)head * N_NODES;
  const float* Mh  = Mrow + (size_t)head * N_NODES;
  const float* dh  = dinv + (size_t)head * N_NODES;
  const _Float16* V = Vbase + (size_t)head * vHeadStride;

  if (tid < 128) { s2s[tid] = s2h[i0 + tid]; Ms[tid] = Mh[i0 + tid]; ds[tid] = dh[i0 + tid]; }
  __syncthreads();

  v8f acc[4] = {};
  const int pr  = tid >> 1;        // P-tile row this thread fills
  const int pcb = (tid & 1) * 16;  // 16-wide column half

  for (int j0 = 0; j0 < N_NODES; j0 += 32) {
    // Stage V tile 32x64 (f16, row stride 64 in global)
    {
      int r = tid >> 3, col = (tid & 7) * 8;
      *(v8h*)&Vs[r][col] = *(const v8h*)(V + (size_t)(j0 + r) * NHID + col);
    }
    // Regenerate P tile 128x32 in f16: p = mask ? exp(lrelu(s1_j+s2_i)-M_i) : 0
    {
      const float s2i = s2s[pr], Mi = Ms[pr];
      const int*   arow = adj + (size_t)(i0 + pr) * N_NODES + j0 + pcb;
      const float* s1p  = s1h + j0 + pcb;
#pragma unroll
      for (int c = 0; c < 16; ++c) {
        float e = lrelu(s1p[c] + s2i);
        float p = (arow[c] > 0) ? __expf(e - Mi) : 0.f;
        Ps[pr][pcb + c] = (_Float16)p;
      }
    }
    __syncthreads();

    v16h a = load_A_frag(&Ps[wave * 16 + mrow][0], halfl);
#pragma unroll
    for (int nt = 0; nt < 4; ++nt) {
      v16h b = load_B_frag(&Vs[0][nt * 16 + mrow], halfl);
      acc[nt] = __builtin_amdgcn_wmma_f32_16x16x32_f16(
          false, a, false, b, (short)0, acc[nt], false, false);
    }
    __syncthreads();
  }

  // Normalize by softmax denominator and store
#pragma unroll
  for (int nt = 0; nt < 4; ++nt)
#pragma unroll
    for (int r = 0; r < 8; ++r) {
      int   row = wave * 16 + 8 * halfl + r;
      float val = acc[nt][r] * ds[row];
      size_t idx = (size_t)(i0 + row) * ldOut + (size_t)head * colHeadStride +
                   nt * 16 + mrow;
      if (F32OUT) ((float*)outv)[idx] = val;
      else        ((_Float16*)outv)[idx] = (_Float16)val;
    }
}

// ---------------------------------------------------------------------------
// Host-side orchestration
// ---------------------------------------------------------------------------
extern "C" void kernel_launch(void* const* d_in, const int* in_sizes, int n_in,
                              void* d_out, int out_size, void* d_ws, size_t ws_size,
                              hipStream_t stream) {
  const float* x   = (const float*)d_in[0];
  const int*   adj = (const int*)d_in[1];
  const float* W   = (const float*)d_in[2];
  const float* a1  = (const float*)d_in[3];
  const float* a2  = (const float*)d_in[4];
  const float* Wf  = (const float*)d_in[5];
  const float* a1f = (const float*)d_in[6];
  const float* a2f = (const float*)d_in[7];
  float* out = (float*)d_out;

  // Carve workspace (256B aligned slices)
  char* w = (char*)d_ws;
  auto carve = [&](size_t bytes) {
    char* p = w;
    w += (bytes + 255) & ~(size_t)255;
    return p;
  };
  _Float16* xh  = (_Float16*)carve((size_t)N_NODES * NFEAT * 2);
  _Float16* Wh  = (_Float16*)carve((size_t)NHEADS * NFEAT * NHID * 2);
  _Float16* Wfh = (_Float16*)carve((size_t)NHEADS * NHID * NCLASS * 2);
  _Float16* h1  = (_Float16*)carve((size_t)NHEADS * N_NODES * NHID * 2);
  _Float16* xc  = (_Float16*)carve((size_t)N_NODES * NHEADS * NHID * 2);
  _Float16* h2  = (_Float16*)carve((size_t)N_NODES * NCLASS * 2);
  float* s1  = (float*)carve((size_t)NHEADS * N_NODES * 4);
  float* s2  = (float*)carve((size_t)NHEADS * N_NODES * 4);
  float* Mr  = (float*)carve((size_t)NHEADS * N_NODES * 4);
  float* di  = (float*)carve((size_t)NHEADS * N_NODES * 4);
  float* s1f = (float*)carve((size_t)N_NODES * 4);
  float* s2f = (float*)carve((size_t)N_NODES * 4);
  float* Mf  = (float*)carve((size_t)N_NODES * 4);
  float* dif = (float*)carve((size_t)N_NODES * 4);
  _Float16* xe = xh;  // xh is dead after gemm1; reuse for ELU output

  // 1) f32 -> f16 conversions
  cvt_f32_to_f16<<<512, 256, 0, stream>>>(x, xh, N_NODES * NFEAT);
  cvt_f32_to_f16<<<256, 256, 0, stream>>>(W, Wh, NHEADS * NFEAT * NHID);
  cvt_f32_to_f16<<<64, 256, 0, stream>>>(Wf, Wfh, NHEADS * NHID * NCLASS);

  // 2) Layer-1 feature GEMM: h1[head] = x @ W[head]
  gemm_wmma<<<dim3(N_NODES / 128, NHEADS), 256, 0, stream>>>(
      xh, NFEAT, NFEAT, Wh, (long)NFEAT * NHID, h1, (long)N_NODES * NHID);

  // 3) Attention projections + per-row softmax stats
  scores_kernel<<<dim3(N_NODES, NHEADS), 64, 0, stream>>>(h1, a1, a2, s1, s2);
  rowstats_kernel<<<dim3(N_NODES, NHEADS), 32, 0, stream>>>(adj, s1, s2, Mr, di);

  // 4) Layer-1 aggregation: xc[:, head*64:(head+1)*64] = softmax(P) @ h1[head]
  agg_wmma<false><<<dim3(N_NODES / 128, NHEADS), 256, 0, stream>>>(
      adj, s1, s2, Mr, di, h1, (long)N_NODES * NHID, xc, NHEADS * NHID, NHID);

  // 5) ELU, layer-2 feature GEMM, projections, stats, final aggregation
  elu_f16<<<512, 256, 0, stream>>>(xc, xe, N_NODES * NHEADS * NHID);
  gemm_wmma<<<dim3(N_NODES / 128, 1), 256, 0, stream>>>(
      xe, NHEADS * NHID, NHEADS * NHID, Wfh, 0, h2, 0);
  scores_kernel<<<dim3(N_NODES, 1), 64, 0, stream>>>(h2, a1f, a2f, s1f, s2f);
  rowstats_kernel<<<dim3(N_NODES, 1), 32, 0, stream>>>(adj, s1f, s2f, Mf, dif);
  agg_wmma<true><<<dim3(N_NODES / 128, 1), 256, 0, stream>>>(
      adj, s1f, s2f, Mf, dif, h2, 0, out, NCLASS, 0);
}
